// PointerNet_70858370449442
// MI455X (gfx1250) — matrix operand order repaired and checked by reading, not compiled, not run on hardware
//
#include <hip/hip_runtime.h>
#include <hip/hip_bf16.h>
#include <math.h>

#define B_DIM 32
#define T_DIM 8192
#define I_DIM 256
#define S_DIM 256
#define H_DIM 256
#define CH    32          // t-chunks for the p-weighted sum
#define NEGV  1e30f

typedef __bf16 v16bf __attribute__((ext_vector_type(16)));
typedef float  v8f   __attribute__((ext_vector_type(8)));

// ---------------- kernel: convert Wi = W1[:, S:] to bf16 ----------------
__global__ void convW_k(const float* __restrict__ W1, __bf16* __restrict__ WiBf) {
  int idx = blockIdx.x * 256 + threadIdx.x;   // 0 .. H*I-1
  int h = idx >> 8, i = idx & 255;
  WiBf[idx] = (__bf16)W1[h * (S_DIM + I_DIM) + S_DIM + i];
}

// ---------------- kernel: sW[b,h] = sum_s state[b,s] * W1[h, s]  (Ws part) ----------------
__global__ void statew_k(const float* __restrict__ st, const float* __restrict__ W1,
                         float* __restrict__ sW) {
  int b = blockIdx.x, h = threadIdx.x;
  __shared__ float ss[S_DIM];
  ss[h] = st[b * S_DIM + h];
  __syncthreads();
  const float* wr = W1 + (size_t)h * (S_DIM + I_DIM);
  float a = 0.f;
#pragma unroll 8
  for (int k = 0; k < S_DIM; ++k) a = fmaf(ss[k], wr[k], a);
  sW[b * H_DIM + h] = a;
}

// ---------------- main fused kernel ----------------
// Per block: 64 rows of x, full H=256 columns.
// 8 waves laid out as 4 (M) x 2 (N): each wave owns a 16x128 tile = 8 WMMA accumulators.
// Epilogue: att[m] = sum_h W2[h]*tanh(G[m,h] + sW[b,h]); logit = att - NEG*(1-mask).
__global__ __launch_bounds__(256)
void pointer_pass_k(const float* __restrict__ x, const __bf16* __restrict__ WiBf,
                    const float* __restrict__ sW, const float* __restrict__ W2,
                    const float* __restrict__ mask, float* __restrict__ out_logit) {
  const int tid    = threadIdx.x;
  const int wave   = tid >> 5;          // 0..7
  const int lane   = tid & 31;
  const int lanelo = lane & 15;
  const int hi     = lane >> 4;         // 0/1
  const int wm     = wave >> 1;         // 0..3  (M direction)
  const int wn     = wave & 1;          // 0..1  (N direction)
  const int M0     = blockIdx.x * 64;
  const int bidx   = M0 >> 13;          // row / T  (64 | 8192, so constant per block)
  const int n0     = wn * 128;

  __shared__ float att_buf[64][33];     // padded against bank conflicts

  v8f acc[8];
#pragma unroll
  for (int t = 0; t < 8; ++t) acc[t] = (v8f){0, 0, 0, 0, 0, 0, 0, 0};

  // A-fragment addressing (16x32 bf16 A, contiguous k order, same as B)
  const int   rowg = M0 + wm * 16 + lanelo;
  const int   kbA  = hi ? 16 : 0;       // lanes 16-31 hold K+16..K+31
  const int   kbB  = hi ? 16 : 0;       // B: lanes 16-31 hold K+16..K+31
  const float* xrow = x + (size_t)rowg * I_DIM;

  for (int k = 0; k < I_DIM; k += 32) {
    // load 16 f32 of this row's K-slice, convert to bf16 fragment in-register
    float4 f0 = *(const float4*)(xrow + k + kbA);
    float4 f1 = *(const float4*)(xrow + k + kbA + 4);
    float4 f2 = *(const float4*)(xrow + k + kbA + 8);
    float4 f3 = *(const float4*)(xrow + k + kbA + 12);
    v16bf a;
    a[0]  = (__bf16)f0.x; a[1]  = (__bf16)f0.y; a[2]  = (__bf16)f0.z; a[3]  = (__bf16)f0.w;
    a[4]  = (__bf16)f1.x; a[5]  = (__bf16)f1.y; a[6]  = (__bf16)f1.z; a[7]  = (__bf16)f1.w;
    a[8]  = (__bf16)f2.x; a[9]  = (__bf16)f2.y; a[10] = (__bf16)f2.z; a[11] = (__bf16)f2.w;
    a[12] = (__bf16)f3.x; a[13] = (__bf16)f3.y; a[14] = (__bf16)f3.z; a[15] = (__bf16)f3.w;

#pragma unroll
    for (int t = 0; t < 8; ++t) {
      // B-fragment: column (= Wi row) n0+t*16+lanelo, 16 contiguous bf16 along K
      const int colg = n0 + t * 16 + lanelo;
      v16bf bf = *(const v16bf*)(WiBf + (size_t)colg * I_DIM + k + kbB);
      acc[t] = __builtin_amdgcn_wmma_f32_16x16x32_bf16(
          /*neg_a=*/false, a, /*neg_b=*/false, bf,
          /*c_mod=*/(short)0, acc[t], /*reuse_a=*/false, /*reuse_b=*/false);
    }
  }

  // fused epilogue: per-lane partial of sum_h W2[h]*tanh(G + sW[b,h]) over its 8 columns
  float contrib[8];
#pragma unroll
  for (int r = 0; r < 8; ++r) contrib[r] = 0.f;
#pragma unroll
  for (int t = 0; t < 8; ++t) {
    const int colg = n0 + t * 16 + lanelo;
    const float w2v = W2[colg];
    const float sv  = sW[bidx * H_DIM + colg];
#pragma unroll
    for (int r = 0; r < 8; ++r)
      contrib[r] += w2v * tanhf(acc[t][r] + sv);
  }
  // C/D layout: vgpr r holds M = r + 8*hi (relative), N = lanelo
#pragma unroll
  for (int r = 0; r < 8; ++r)
    att_buf[wm * 16 + hi * 8 + r][wn * 16 + lanelo] = contrib[r];
  __syncthreads();

  if (tid < 64) {
    float s = 0.f;
#pragma unroll
    for (int j = 0; j < 32; ++j) s += att_buf[tid][j];
    const int m = M0 + tid;
    out_logit[m] = s - NEGV * (1.0f - mask[m]);
  }
}

// ---------------- softmax over T per batch ----------------
__global__ void softmax_k(const float* __restrict__ logit, float* __restrict__ p) {
  int b = blockIdx.x, tid = threadIdx.x;
  __shared__ float red[256];
  const float* row = logit + (size_t)b * T_DIM;
  float m = -3.0e38f;
  for (int t = tid; t < T_DIM; t += 256) m = fmaxf(m, row[t]);
  red[tid] = m; __syncthreads();
  for (int s = 128; s > 0; s >>= 1) { if (tid < s) red[tid] = fmaxf(red[tid], red[tid + s]); __syncthreads(); }
  m = red[0]; __syncthreads();
  float sum = 0.f;
  for (int t = tid; t < T_DIM; t += 256) sum += __expf(row[t] - m);
  red[tid] = sum; __syncthreads();
  for (int s = 128; s > 0; s >>= 1) { if (tid < s) red[tid] += red[tid + s]; __syncthreads(); }
  const float inv = 1.0f / red[0];
  for (int t = tid; t < T_DIM; t += 256)
    p[(size_t)b * T_DIM + t] = __expf(row[t] - m) * inv;
}

// ---------------- out[b,i] = sum_t p[b,t]*x[b,t,i], stage 1 (chunked) ----------------
__global__ void wsum1_k(const float* __restrict__ x, const float* __restrict__ p,
                        float* __restrict__ partial) {
  const int i = threadIdx.x;          // 0..255
  const int c = blockIdx.x;           // chunk 0..CH-1
  const int b = blockIdx.y;           // batch
  const int t0 = c * (T_DIM / CH);
  float acc = 0.f;
  for (int t = t0; t < t0 + T_DIM / CH; ++t)
    acc = fmaf(p[(size_t)b * T_DIM + t], x[((size_t)b * T_DIM + t) * I_DIM + i], acc);
  partial[((size_t)c * B_DIM + b) * I_DIM + i] = acc;
}

// ---------------- stage 2: reduce chunks ----------------
__global__ void wsum2_k(const float* __restrict__ partial, float* __restrict__ outv) {
  const int i = threadIdx.x, b = blockIdx.x;
  float acc = 0.f;
#pragma unroll
  for (int c = 0; c < CH; ++c) acc += partial[((size_t)c * B_DIM + b) * I_DIM + i];
  outv[b * I_DIM + i] = acc;
}

// ---------------- GRU cell ----------------
__global__ void gru_k(const float* __restrict__ outv, const float* __restrict__ init,
                      const float* __restrict__ w_ih, const float* __restrict__ w_hh,
                      const float* __restrict__ b_ih, const float* __restrict__ b_hh,
                      float* __restrict__ state) {
  const int b = blockIdx.x, s = threadIdx.x;
  __shared__ float sx[I_DIM], sh[S_DIM];
  sx[s] = outv[b * I_DIM + s];
  sh[s] = init[b * S_DIM + s];
  __syncthreads();
  float gi[3], gh[3];
#pragma unroll
  for (int g = 0; g < 3; ++g) {
    const float* wr = w_ih + (size_t)(g * S_DIM + s) * I_DIM;
    const float* hr = w_hh + (size_t)(g * S_DIM + s) * S_DIM;
    float a1 = b_ih[g * S_DIM + s], a2 = b_hh[g * S_DIM + s];
#pragma unroll 8
    for (int k = 0; k < I_DIM; ++k) { a1 = fmaf(sx[k], wr[k], a1); a2 = fmaf(sh[k], hr[k], a2); }
    gi[g] = a1; gh[g] = a2;
  }
  const float r = 1.0f / (1.0f + __expf(-(gi[0] + gh[0])));
  const float z = 1.0f / (1.0f + __expf(-(gi[1] + gh[1])));
  const float n = tanhf(gi[2] + r * gh[2]);
  state[b * S_DIM + s] = (1.0f - z) * n + z * sh[s];
}

extern "C" void kernel_launch(void* const* d_in, const int* in_sizes, int n_in,
                              void* d_out, int out_size, void* d_ws, size_t ws_size,
                              hipStream_t stream) {
  const float* x    = (const float*)d_in[0];
  const float* init = (const float*)d_in[1];
  const float* mask = (const float*)d_in[2];
  const float* W1   = (const float*)d_in[3];
  const float* W2   = (const float*)d_in[4];
  const float* w_ih = (const float*)d_in[5];
  const float* w_hh = (const float*)d_in[6];
  const float* b_ih = (const float*)d_in[7];
  const float* b_hh = (const float*)d_in[8];

  char* ws = (char*)d_ws;
  __bf16* WiBf   = (__bf16*)(ws);                       // 128 KB
  float*  sW1    = (float*)(ws + (131072));             // 32 KB
  float*  sW2    = (float*)(ws + (131072 + 32768));     // 32 KB
  float*  p      = (float*)(ws + (196608));             // 1 MB
  float*  part   = (float*)(ws + (196608 + 1048576));   // 1 MB
  float*  outv   = (float*)(ws + (196608 + 2097152));   // 32 KB
  float*  state  = (float*)(ws + (196608 + 2097152 + 32768)); // 32 KB

  float* logit1 = (float*)d_out;
  float* logit2 = logit1 + (size_t)B_DIM * T_DIM;

  convW_k <<<(H_DIM * I_DIM) / 256, 256, 0, stream>>>(W1, WiBf);
  statew_k<<<B_DIM, 256, 0, stream>>>(init, W1, sW1);
  pointer_pass_k<<<(B_DIM * T_DIM) / 64, 256, 0, stream>>>(x, WiBf, sW1, W2, mask, logit1);
  softmax_k<<<B_DIM, 256, 0, stream>>>(logit1, p);
  wsum1_k <<<dim3(CH, B_DIM), 256, 0, stream>>>(x, p, part);
  wsum2_k <<<B_DIM, 256, 0, stream>>>(part, outv);
  gru_k   <<<B_DIM, 256, 0, stream>>>(outv, init, w_ih, w_hh, b_ih, b_hh, state);
  statew_k<<<B_DIM, 256, 0, stream>>>(state, W1, sW2);
  pointer_pass_k<<<(B_DIM * T_DIM) / 64, 256, 0, stream>>>(x, WiBf, sW2, W2, mask, logit2);
}
